// Denoiser_31774168056204
// MI455X (gfx1250) — matrix operand order repaired and checked by reading, not compiled, not run on hardware
//
#include <hip/hip_runtime.h>
#include <math.h>
#include <stdint.h>

// ---------------------------------------------------------------------------
// MI455X (gfx1250) implementation of the denoiser reference.
// Dense GEMMs -> v_wmma_f32_16x16x32_f16 (f32 accumulate), double-buffered LDS
// with software-pipelined global->reg->LDS staging.
// Attention K/V staging -> global_load_async_to_lds_* (ASYNCcnt path).
// ---------------------------------------------------------------------------

typedef _Float16 h16;
typedef __attribute__((ext_vector_type(4)))  _Float16 v4h;
typedef __attribute__((ext_vector_type(8)))  _Float16 v8h;
typedef __attribute__((ext_vector_type(16))) _Float16 v16h;
typedef __attribute__((ext_vector_type(8)))  float    v8f;

#define LDT 40              // LDS tile stride in halves (80B -> 16B aligned frags)
#define TT  91              // sequence length
#define NTOK 11648          // 2*64*91
#define ACT_NONE 0
#define ACT_GELU 1
#define ACT_RELU 2

__device__ __forceinline__ float gelu_exact(float x) {
    return 0.5f * x * (1.0f + erff(x * 0.70710678118654752f));
}

// Build a 16-half fragment from two aligned 8-half LDS loads.
#define FRAG16(p)                                                        \
    ({ v8h _lo = *(const v8h*)(p); v8h _hi = *(const v8h*)((p) + 8);     \
       __builtin_shufflevector(_lo, _hi, 0,1,2,3,4,5,6,7,8,9,10,11,12,13,14,15); })

// Async copy helpers (CDNA5 ASYNCcnt path). LDS byte address = low 32 bits of
// the generic pointer (flat->LDS mapping uses addr[31:0]).
__device__ __forceinline__ void async_b32(uint32_t lds, const float* g) {
    asm volatile("global_load_async_to_lds_b32 %0, %1, off"
                 :: "v"(lds), "v"((uint64_t)(uintptr_t)g) : "memory");
}
__device__ __forceinline__ void async_b128(uint32_t lds, const float* g) {
    asm volatile("global_load_async_to_lds_b128 %0, %1, off"
                 :: "v"(lds), "v"((uint64_t)(uintptr_t)g) : "memory");
}
__device__ __forceinline__ void wait_async0() {
    asm volatile("s_wait_asynccnt 0x0" ::: "memory");
}

// ---------------------------------------------------------------------------
// GEMM tile staging, split into phases for software pipelining.
// Fast path requires a full 64x32 tile (K/N interior; all our K are %4==0).
// ---------------------------------------------------------------------------
__device__ __forceinline__ void load_tile_regs(
    const float* __restrict__ A, const float* __restrict__ W,
    float4* fa, float4* fb, int tid, int m0, int n0, int k0, int N, int K)
{
#pragma unroll
    for (int j = 0; j < 4; ++j) {                  // A: 512 float4 chunks
        int c = tid + j * 128;
        int r = c >> 3, kq = (c & 7) << 2;         // 8 chunks per 32-wide row
        fa[j] = *(const float4*)(A + (size_t)(m0 + r) * K + k0 + kq);
    }
#pragma unroll
    for (int j = 0; j < 4; ++j) {                  // B: 512 float4 chunks
        int c = tid + j * 128;
        int kk = c >> 4, nq = (c & 15) << 2;       // 16 chunks per 64-wide row
        fb[j] = *(const float4*)(W + (size_t)(k0 + kk) * N + n0 + nq);
    }
}

__device__ __forceinline__ void store_tile_lds(
    const float4* fa, const float4* fb, h16* sA, h16* sB, int tid)
{
#pragma unroll
    for (int j = 0; j < 4; ++j) {
        int c = tid + j * 128;
        int r = c >> 3, kq = (c & 7) << 2;
        v4h hv = {(h16)fa[j].x, (h16)fa[j].y, (h16)fa[j].z, (h16)fa[j].w};
        *(v4h*)(sA + r * LDT + kq) = hv;           // b64 ds store, 8B aligned
    }
#pragma unroll
    for (int j = 0; j < 4; ++j) {
        int c = tid + j * 128;
        int kk = c >> 4, nq = (c & 15) << 2;
        sB[(nq + 0) * LDT + kk] = (h16)fb[j].x;    // transposed scatter
        sB[(nq + 1) * LDT + kk] = (h16)fb[j].y;
        sB[(nq + 2) * LDT + kk] = (h16)fb[j].z;
        sB[(nq + 3) * LDT + kk] = (h16)fb[j].w;
    }
}

// Guarded scalar staging for edge tiles (K=20, K%32 tail, N=1028/3 edges).
__device__ __forceinline__ void stage_slow(
    const float* __restrict__ A, const float* __restrict__ W,
    h16* sA, h16* sB, int tid, int m0, int n0, int k0, int N, int K)
{
    for (int i = tid; i < 64 * 32; i += 128) {
        int r = i >> 5, kk = i & 31;
        float va = (k0 + kk < K) ? A[(size_t)(m0 + r) * K + (k0 + kk)] : 0.0f;
        sA[r * LDT + kk] = (h16)va;
    }
    for (int i = tid; i < 64 * 32; i += 128) {
        int kk = i >> 6, n = i & 63;
        float vb = (k0 + kk < K && n0 + n < N)
                       ? W[(size_t)(k0 + kk) * N + (n0 + n)] : 0.0f;
        sB[n * LDT + kk] = (h16)vb;
    }
}

// ---------------------------------------------------------------------------
// C[M,N] = act(A[M,K] @ W[K,N] + bias) (+ R)   -- M must be a multiple of 64.
// 128 threads (4 waves). Each wave: 16 rows x 64 cols via 4 WMMA accumulators.
// Double-buffered LDS; next tile's global loads issued before this tile's
// WMMAs, converted/stored after them (hides VMEM latency under the WMMA pipe).
// ---------------------------------------------------------------------------
__global__ __launch_bounds__(128) void gemm_wmma(
    const float* __restrict__ A, const float* __restrict__ W,
    const float* __restrict__ bias, const float* R, float* C,
    int M, int N, int K, int act)
{
    __shared__ alignas(16) h16 lA[2][64 * LDT];   // [row][k]
    __shared__ alignas(16) h16 lB[2][64 * LDT];   // [n][k] (transposed weights)

    const int tid  = threadIdx.x;
    const int wave = tid >> 5;
    const int lane = tid & 31;
    const int hilo = lane >> 4;        // 0: lanes 0-15, 1: lanes 16-31
    const int l15  = lane & 15;
    const int m0   = blockIdx.y * 64;
    const int n0   = blockIdx.x * 64;
    const bool nfull = (n0 + 64 <= N);

    v8f acc0 = {}, acc1 = {}, acc2 = {}, acc3 = {};

    // Prologue: stage first tile.
    if (nfull && 32 <= K) {
        float4 fa[4], fb[4];
        load_tile_regs(A, W, fa, fb, tid, m0, n0, 0, N, K);
        store_tile_lds(fa, fb, lA[0], lB[0], tid);
    } else {
        stage_slow(A, W, lA[0], lB[0], tid, m0, n0, 0, N, K);
    }
    __syncthreads();

    int p = 0;
    for (int k0 = 0; k0 < K; k0 += 32) {
        const int  kn       = k0 + 32;
        const bool havenext = kn < K;
        const bool nextfast = havenext && nfull && (kn + 32 <= K);

        float4 fa[4], fb[4];
        if (nextfast)       // issue next tile's global loads NOW (consumed later)
            load_tile_regs(A, W, fa, fb, tid, m0, n0, kn, N, K);
        else if (havenext)  // rare edge tile: scalar staging up front
            stage_slow(A, W, lA[p ^ 1], lB[p ^ 1], tid, m0, n0, kn, N, K);

        // Fragments: A row = 16*wave + l15 (lanes 0-15: K0-15,
        // lanes 16-31: K16-31); B col = nt*16 + l15 (K0-15 / K16-31).
        const h16* ap = &lA[p][(16 * wave + l15) * LDT + hilo * 16];
        v8h alo = *(const v8h*)ap;
        v8h ahi = *(const v8h*)(ap + 8);
        v16h af = __builtin_shufflevector(alo, ahi,
                     0,1,2,3,4,5,6,7,8,9,10,11,12,13,14,15);
        v16h bf0 = FRAG16(&lB[p][( 0 + l15) * LDT + hilo * 16]);
        v16h bf1 = FRAG16(&lB[p][(16 + l15) * LDT + hilo * 16]);
        v16h bf2 = FRAG16(&lB[p][(32 + l15) * LDT + hilo * 16]);
        v16h bf3 = FRAG16(&lB[p][(48 + l15) * LDT + hilo * 16]);

        acc0 = __builtin_amdgcn_wmma_f32_16x16x32_f16(false, af, false, bf0,
                                                      (short)0, acc0, false, false);
        acc1 = __builtin_amdgcn_wmma_f32_16x16x32_f16(false, af, false, bf1,
                                                      (short)0, acc1, false, false);
        acc2 = __builtin_amdgcn_wmma_f32_16x16x32_f16(false, af, false, bf2,
                                                      (short)0, acc2, false, false);
        acc3 = __builtin_amdgcn_wmma_f32_16x16x32_f16(false, af, false, bf3,
                                                      (short)0, acc3, false, false);

        if (nextfast)       // convert + LDS-store the already-loaded next tile
            store_tile_lds(fa, fb, lA[p ^ 1], lB[p ^ 1], tid);
        __syncthreads();
        p ^= 1;
    }

    // Store: C/D layout -> VGPR i, lanes 0-15: M=i, N=lane; lanes 16-31: M=8+i.
    const int mrow = m0 + 16 * wave + hilo * 8;
    v8f accs[4] = {acc0, acc1, acc2, acc3};
#pragma unroll
    for (int nt = 0; nt < 4; ++nt) {
        int n = n0 + nt * 16 + l15;
        if (n < N) {
            float bv = bias[n];
#pragma unroll
            for (int i = 0; i < 8; ++i) {
                float val = accs[nt][i] + bv;
                if (act == ACT_GELU)      val = gelu_exact(val);
                else if (act == ACT_RELU) val = fmaxf(val, 0.0f);
                size_t idx = (size_t)(mrow + i) * N + n;
                if (R) val += R[idx];
                C[idx] = val;
            }
        }
    }
}

// ---------------------------------------------------------------------------
// LayerNorm over D=256, wave-per-row (wave32: 8 elements / lane).
// ---------------------------------------------------------------------------
__global__ __launch_bounds__(256) void layernorm_k(
    const float* __restrict__ x, const float* __restrict__ g,
    const float* __restrict__ b, float* __restrict__ y, int rows)
{
    int wave = threadIdx.x >> 5, lane = threadIdx.x & 31;
    int row = blockIdx.x * 8 + wave;
    if (row >= rows) return;
    const float* xr = x + (size_t)row * 256;
    float v[8], s = 0.0f;
#pragma unroll
    for (int j = 0; j < 8; ++j) { v[j] = xr[lane + 32 * j]; s += v[j]; }
#pragma unroll
    for (int o = 16; o; o >>= 1) s += __shfl_xor(s, o, 32);
    float mu = s * (1.0f / 256.0f);
    float s2 = 0.0f;
#pragma unroll
    for (int j = 0; j < 8; ++j) { float d = v[j] - mu; s2 += d * d; }
#pragma unroll
    for (int o = 16; o; o >>= 1) s2 += __shfl_xor(s2, o, 32);
    float inv = rsqrtf(s2 * (1.0f / 256.0f) + 1e-5f);
    float* yr = y + (size_t)row * 256;
#pragma unroll
    for (int j = 0; j < 8; ++j) {
        int c = lane + 32 * j;
        yr[c] = (v[j] - mu) * inv * g[c] + b[c];
    }
}

// ---------------------------------------------------------------------------
// Self-attention core. grid = (b*64+a)*4+h blocks (512), 128 threads.
// K/V staged via async global->LDS DMA: K with +1 padding (row reads in the
// score loop -> conflict-free), V unpadded (column reads -> conflict-free)
// so V rows stay 16B aligned for async_b128.
// ---------------------------------------------------------------------------
__global__ __launch_bounds__(128) void attn_self(
    const float* __restrict__ q, const float* __restrict__ k,
    const float* __restrict__ v, const int* __restrict__ fmask,
    float* __restrict__ o)
{
    __shared__ float Kl[TT][65];
    __shared__ float Vl[TT][64];
    __shared__ float qrow[64];
    __shared__ float wsum[96];
    __shared__ float red[128];
    __shared__ int   mk[96];
    __shared__ int   anyv;

    int tid = threadIdx.x;
    int h = blockIdx.x & 3;
    int ba = blockIdx.x >> 2;                 // b*64 + a
    size_t base = (size_t)ba * TT * 256 + h * 64;

    uint32_t ldsK = (uint32_t)(uintptr_t)(&Kl[0][0]);
    uint32_t ldsV = (uint32_t)(uintptr_t)(&Vl[0][0]);
    for (int i = tid; i < TT * 64; i += 128) {
        int t = i >> 6, d = i & 63;
        async_b32(ldsK + (uint32_t)(t * 65 + d) * 4, k + base + (size_t)t * 256 + d);
    }
    for (int i = tid; i < TT * 16; i += 128) {
        int t = i >> 4, c = (i & 15) << 2;
        async_b128(ldsV + (uint32_t)(t * 64 + c) * 4, v + base + (size_t)t * 256 + c);
    }
    if (tid < 96) mk[tid] = (tid < TT) ? (fmask[ba * TT + tid] != 0) : 0;
    if (tid == 0) anyv = 0;
    wait_async0();
    __syncthreads();
    if (tid < TT && mk[tid]) atomicOr(&anyv, 1);
    __syncthreads();

    for (int t = 0; t < TT; ++t) {
        if (tid < 64) qrow[tid] = q[base + (size_t)t * 256 + tid];
        __syncthreads();
        float sc = -1e30f;
        if (tid < TT) {
            float a = 0.0f;
            for (int d = 0; d < 64; ++d) a += qrow[d] * Kl[tid][d];
            sc = mk[tid] ? a * 0.125f : -1e30f;
        }
        red[tid] = sc;
        __syncthreads();
        for (int s = 64; s; s >>= 1) {
            if (tid < s) red[tid] = fmaxf(red[tid], red[tid + s]);
            __syncthreads();
        }
        float mx = red[0];
        __syncthreads();
        float e = (tid < TT) ? __expf(sc - mx) : 0.0f;
        red[tid] = e;
        __syncthreads();
        for (int s = 64; s; s >>= 1) {
            if (tid < s) red[tid] += red[tid + s];
            __syncthreads();
        }
        float den = red[0];
        if (tid < TT) wsum[tid] = (mk[tid] && anyv) ? e / den : 0.0f;
        __syncthreads();
        if (tid < 64) {
            float a = 0.0f;
            for (int s2 = 0; s2 < TT; ++s2) a += wsum[s2] * Vl[s2][tid];
            o[base + (size_t)t * 256 + tid] = a;
        }
        __syncthreads();
    }
}

// ---------------------------------------------------------------------------
// Cross-attention core (keys = agents at fixed t). grid = (b*91+t)*4+h (728).
// ---------------------------------------------------------------------------
__global__ __launch_bounds__(128) void attn_cross(
    const float* __restrict__ q, const float* __restrict__ k,
    const float* __restrict__ v, const int* __restrict__ fmask,
    float* __restrict__ o)
{
    __shared__ float Kl[64][65];
    __shared__ float Vl[64][64];
    __shared__ float qrow[64];
    __shared__ float wsum[64];
    __shared__ float red[128];
    __shared__ int   mk[64];
    __shared__ int   anyv;

    int tid = threadIdx.x;
    int h = blockIdx.x & 3;
    int bt = blockIdx.x >> 2;
    int b = bt / TT, t = bt % TT;

    uint32_t ldsK = (uint32_t)(uintptr_t)(&Kl[0][0]);
    uint32_t ldsV = (uint32_t)(uintptr_t)(&Vl[0][0]);
    for (int i = tid; i < 64 * 64; i += 128) {
        int s = i >> 6, d = i & 63;
        size_t kb = ((size_t)(b * 64 + s) * TT + t) * 256 + h * 64 + d;
        async_b32(ldsK + (uint32_t)(s * 65 + d) * 4, k + kb);
    }
    for (int i = tid; i < 64 * 16; i += 128) {
        int s = i >> 4, c = (i & 15) << 2;
        size_t kb = ((size_t)(b * 64 + s) * TT + t) * 256 + h * 64 + c;
        async_b128(ldsV + (uint32_t)(s * 64 + c) * 4, v + kb);
    }
    if (tid < 64) mk[tid] = (fmask[(b * 64 + tid) * TT + t] != 0);
    if (tid == 0) anyv = 0;
    wait_async0();
    __syncthreads();
    if (tid < 64 && mk[tid]) atomicOr(&anyv, 1);
    __syncthreads();

    for (int a = 0; a < 64; ++a) {
        size_t qb = ((size_t)(b * 64 + a) * TT + t) * 256 + h * 64;
        if (tid < 64) qrow[tid] = q[qb + tid];
        __syncthreads();
        float sc = -1e30f;
        if (tid < 64) {
            float acc = 0.0f;
            for (int d = 0; d < 64; ++d) acc += qrow[d] * Kl[tid][d];
            sc = mk[tid] ? acc * 0.125f : -1e30f;
        }
        red[tid] = sc;
        __syncthreads();
        for (int s = 64; s; s >>= 1) {
            if (tid < s) red[tid] = fmaxf(red[tid], red[tid + s]);
            __syncthreads();
        }
        float mx = red[0];
        __syncthreads();
        float e = (tid < 64) ? __expf(sc - mx) : 0.0f;
        red[tid] = e;
        __syncthreads();
        for (int s = 64; s; s >>= 1) {
            if (tid < s) red[tid] += red[tid + s];
            __syncthreads();
        }
        float den = red[0];
        if (tid < 64) wsum[tid] = (mk[tid] && anyv) ? e / den : 0.0f;
        __syncthreads();
        if (tid < 64) {
            float acc = 0.0f;
            for (int s2 = 0; s2 < 64; ++s2) acc += wsum[s2] * Vl[s2][tid];
            o[qb + tid] = acc;
        }
        __syncthreads();
    }
}

// ---------------------------------------------------------------------------
// Road attention core: 512 roadgraph keys (L2-hot). grid = ((b*4)+h)*64+a
// (512 blocks), 256 threads. k/v: [B,512,256]. Prefetch hides L2 latency.
// ---------------------------------------------------------------------------
__global__ __launch_bounds__(256) void attn_road(
    const float* __restrict__ q, const float* __restrict__ k,
    const float* __restrict__ v, const int* __restrict__ rmask,
    float* __restrict__ o)
{
    __shared__ float ws2[512];
    __shared__ float red[256];
    __shared__ float qrow[64];
    __shared__ int   mk[512];
    __shared__ int   anyv;

    int tid = threadIdx.x;
    int a = blockIdx.x & 63;
    int bh = blockIdx.x >> 6;
    int h = bh & 3, b = bh >> 2;
    size_t kbase = (size_t)b * 512 * 256 + h * 64;

    for (int i = tid; i < 512; i += 256) mk[i] = (rmask[b * 512 + i] != 0);
    if (tid == 0) anyv = 0;
    __syncthreads();
    if (mk[tid] || mk[tid + 256]) atomicOr(&anyv, 1);
    __syncthreads();

    for (int t = 0; t < TT; ++t) {
        size_t qb = ((size_t)(b * 64 + a) * TT + t) * 256 + h * 64;
        if (tid < 64) qrow[tid] = q[qb + tid];
        __syncthreads();
        for (int r = tid; r < 512; r += 256) {
            const float* kr = k + kbase + (size_t)r * 256;
            float acc = 0.0f;
            for (int d = 0; d < 64; ++d) acc += qrow[d] * kr[d];
            ws2[r] = mk[r] ? acc * 0.125f : -1e30f;
        }
        __syncthreads();
        red[tid] = fmaxf(ws2[tid], ws2[tid + 256]);
        __syncthreads();
        for (int s = 128; s; s >>= 1) {
            if (tid < s) red[tid] = fmaxf(red[tid], red[tid + s]);
            __syncthreads();
        }
        float mx = red[0];
        __syncthreads();
        float e0 = __expf(ws2[tid] - mx);
        float e1 = __expf(ws2[tid + 256] - mx);
        red[tid] = e0 + e1;
        __syncthreads();
        for (int s = 128; s; s >>= 1) {
            if (tid < s) red[tid] += red[tid + s];
            __syncthreads();
        }
        float inv = anyv ? (1.0f / red[0]) : 0.0f;
        __syncthreads();
        ws2[tid] = e0 * inv;
        ws2[tid + 256] = e1 * inv;
        __syncthreads();
        if (tid < 64) {
            float acc = 0.0f;
            for (int r = 0; r < 512; ++r) {
                if (r + 8 < 512)
                    __builtin_prefetch(v + kbase + (size_t)(r + 8) * 256 + tid, 0, 1);
                acc += ws2[r] * v[kbase + (size_t)r * 256 + tid];
            }
            o[qb + tid] = acc;
        }
        __syncthreads();
    }
}

// ---------------------------------------------------------------------------
// Host-side orchestration
// ---------------------------------------------------------------------------
static inline void gemm(hipStream_t s, const float* A, const float* W,
                        const float* bias, const float* R, float* C,
                        int M, int N, int K, int act) {
    dim3 grid((N + 63) / 64, M / 64);
    gemm_wmma<<<grid, 128, 0, s>>>(A, W, bias, R, C, M, N, K, act);
}

extern "C" void kernel_launch(void* const* d_in, const int* in_sizes, int n_in,
                              void* d_out, int out_size, void* d_ws, size_t ws_size,
                              hipStream_t stream) {
    (void)in_sizes; (void)n_in; (void)out_size; (void)ws_size;
    const float* noisy = (const float*)d_in[0];
    const float* rg_in = (const float*)d_in[1];   // [2,512,10,2] == [1024,20]
    // feature MLP params
    const float *fW1 = (const float*)d_in[2],  *fb1 = (const float*)d_in[3];
    const float *fW2 = (const float*)d_in[4],  *fb2 = (const float*)d_in[5];
    const float *fW3 = (const float*)d_in[6],  *fb3 = (const float*)d_in[7];
    const float *fWo = (const float*)d_in[8],  *fbo = (const float*)d_in[9];
    const float *fWs = (const float*)d_in[10], *fbs = (const float*)d_in[11];
    // rg MLP params
    const float *rW1 = (const float*)d_in[12], *rb1 = (const float*)d_in[13];
    const float *rW2 = (const float*)d_in[14], *rb2 = (const float*)d_in[15];
    const float *rW3 = (const float*)d_in[16], *rb3 = (const float*)d_in[17];
    // out MLP params
    const float *oW1 = (const float*)d_in[18], *ob1 = (const float*)d_in[19];
    const float *oW2 = (const float*)d_in[20], *ob2 = (const float*)d_in[21];
    const float *oW3 = (const float*)d_in[22], *ob3 = (const float*)d_in[23];
    // attention params [15,...]
    const float *aWq = (const float*)d_in[24], *abq = (const float*)d_in[25];
    const float *aWk = (const float*)d_in[26], *abk = (const float*)d_in[27];
    const float *aWv = (const float*)d_in[28], *abv = (const float*)d_in[29];
    const float *aWo = (const float*)d_in[30], *abo = (const float*)d_in[31];
    const float *aG  = (const float*)d_in[32], *aB  = (const float*)d_in[33];
    // ffn params [15,...]
    const float *gG  = (const float*)d_in[34], *gB  = (const float*)d_in[35];
    const float *gW1 = (const float*)d_in[36], *gb1 = (const float*)d_in[37];
    const float *gW2 = (const float*)d_in[38], *gb2 = (const float*)d_in[39];
    const float *gW3 = (const float*)d_in[40], *gb3 = (const float*)d_in[41];
    const int* fmask = (const int*)d_in[42];
    const int* rmask = (const int*)d_in[43];

    // workspace layout (floats)
    float* wsf = (float*)d_ws;
    size_t off = 0;
    float* x    = wsf + off; off += (size_t)NTOK * 256;
    float* xn   = wsf + off; off += (size_t)NTOK * 256;
    float* qb   = wsf + off; off += (size_t)NTOK * 256;
    float* kb   = wsf + off; off += (size_t)NTOK * 256;
    float* vb   = wsf + off; off += (size_t)NTOK * 256;
    float* ob   = wsf + off; off += (size_t)NTOK * 256;
    float* h1   = wsf + off; off += (size_t)NTOK * 1028;
    float* h2   = wsf + off; off += (size_t)NTOK * 1028;
    float* rgA  = wsf + off; off += (size_t)1024 * 256;
    float* rgB  = wsf + off; off += (size_t)1024 * 256;
    float* rgF  = wsf + off; off += (size_t)1024 * 256;

    // ---- feature MLP: h = gelu^3 chain; x = h@Wo+bo + noisy@Ws+bs ----
    gemm(stream, noisy, fW1, fb1, nullptr, h1, NTOK, 1024, 1408, ACT_GELU);
    gemm(stream, h1,    fW2, fb2, nullptr, h2, NTOK, 1024, 1024, ACT_GELU);
    gemm(stream, h2,    fW3, fb3, nullptr, h1, NTOK, 1024, 1024, ACT_GELU);
    gemm(stream, noisy, fWs, fbs, nullptr, x,  NTOK, 256, 1408, ACT_NONE);
    gemm(stream, h1,    fWo, fbo, x,       x,  NTOK, 256, 1024, ACT_NONE);

    // ---- roadgraph MLP: [1024,20] -> 256 relu -> 256 relu -> 256 ----
    gemm(stream, rg_in, rW1, rb1, nullptr, rgA, 1024, 256, 20,  ACT_RELU);
    gemm(stream, rgA,   rW2, rb2, nullptr, rgB, 1024, 256, 256, ACT_RELU);
    gemm(stream, rgB,   rW3, rb3, nullptr, rgF, 1024, 256, 256, ACT_NONE);

    // ---- transformer blocks ----
    static const char BL[31] = "sfsfcfsfsfrfsfcfsfrfsfsfcfsfsf";
    int ia = 0, iff = 0;
    for (int i = 0; i < 30; ++i) {
        char c = BL[i];
        if (c == 'f') {
            layernorm_k<<<NTOK / 8, 256, 0, stream>>>(x, gG + iff * 256,
                                                      gB + iff * 256, xn, NTOK);
            gemm(stream, xn, gW1 + (size_t)iff * 256 * 1028, gb1 + iff * 1028,
                 nullptr, h1, NTOK, 1028, 256, ACT_GELU);
            gemm(stream, h1, gW2 + (size_t)iff * 1028 * 1028, gb2 + iff * 1028,
                 nullptr, h2, NTOK, 1028, 1028, ACT_GELU);
            gemm(stream, h2, gW3 + (size_t)iff * 1028 * 256, gb3 + iff * 256,
                 x, x, NTOK, 256, 1028, ACT_NONE);
            ++iff;
        } else {
            const float* Wq = aWq + (size_t)ia * 256 * 256;
            const float* Wk = aWk + (size_t)ia * 256 * 256;
            const float* Wv = aWv + (size_t)ia * 256 * 256;
            const float* Wo = aWo + (size_t)ia * 256 * 256;
            const float* bq = abq + ia * 256;
            const float* bk = abk + ia * 256;
            const float* bv = abv + ia * 256;
            const float* bo = abo + ia * 256;
            layernorm_k<<<NTOK / 8, 256, 0, stream>>>(x, aG + ia * 256,
                                                      aB + ia * 256, xn, NTOK);
            gemm(stream, xn, Wq, bq, nullptr, qb, NTOK, 256, 256, ACT_NONE);
            if (c == 'r') {
                gemm(stream, rgF, Wk, bk, nullptr, kb, 1024, 256, 256, ACT_NONE);
                gemm(stream, rgF, Wv, bv, nullptr, vb, 1024, 256, 256, ACT_NONE);
                attn_road<<<512, 256, 0, stream>>>(qb, kb, vb, rmask, ob);
            } else {
                gemm(stream, xn, Wk, bk, nullptr, kb, NTOK, 256, 256, ACT_NONE);
                gemm(stream, xn, Wv, bv, nullptr, vb, NTOK, 256, 256, ACT_NONE);
                if (c == 's')
                    attn_self<<<512, 128, 0, stream>>>(qb, kb, vb, fmask, ob);
                else
                    attn_cross<<<728, 128, 0, stream>>>(qb, kb, vb, fmask, ob);
            }
            // residual uses the NORMALIZED x (per reference)
            gemm(stream, ob, Wo, bo, xn, x, NTOK, 256, 256, ACT_NONE);
            ++ia;
        }
    }

    // ---- out MLP ----
    gemm(stream, x,  oW1, ob1, nullptr, qb, NTOK, 256, 256, ACT_GELU);
    gemm(stream, qb, oW2, ob2, nullptr, kb, NTOK, 256, 256, ACT_GELU);
    gemm(stream, kb, oW3, ob3, nullptr, (float*)d_out, NTOK, 3, 256, ACT_NONE);
}
